// MoELayer_62775241998544
// MI455X (gfx1250) — matrix operand order repaired and checked by reading, not compiled, not run on hardware
//
#include <hip/hip_runtime.h>
#include <hip/hip_bf16.h>

// ---------------------------------------------------------------------------
// MoE layer for MI455X (gfx1250, wave32, WMMA).
// bf16 inputs / f32 accumulate via v_wmma_f32_16x16x32_bf16.
// Sparse top-2 grouped GEMM (exactly equal to the reference's dense combine).
// Double-buffered LDS pipeline: one barrier per K-step, global loads staged in
// registers and overlapped with WMMA issue.
// ---------------------------------------------------------------------------

typedef __bf16 bf16_t;
typedef bf16_t v16bf __attribute__((ext_vector_type(16)));
typedef float  v8f   __attribute__((ext_vector_type(8)));

#define T_TOK  2048
#define H_DIM  1024
#define I_DIM  2816
#define IS_DIM 1408
#define E_NUM  8

union Frag16 { v16bf v; uint4 q[2]; };

__device__ __forceinline__ v8f v8f_zero() {
  v8f z;
  #pragma unroll
  for (int i = 0; i < 8; ++i) z[i] = 0.f;
  return z;
}

__device__ __forceinline__ v8f wmma_bf16(const Frag16& a, const Frag16& b, v8f c) {
  // D = A(16x32 bf16) * B(32x16 bf16) + C(16x16 f32)
  return __builtin_amdgcn_wmma_f32_16x16x32_bf16(false, a.v, false, b.v,
                                                 (short)0, c, false, false);
}

// pack two floats into adjacent bf16 lanes (k, k+1)
__device__ __forceinline__ unsigned pack_bf16(float lo, float hi) {
  union { bf16_t h[2]; unsigned u; } p;
  p.h[0] = (bf16_t)lo;
  p.h[1] = (bf16_t)hi;
  return p.u;
}

__device__ __forceinline__ float fast_silu(float g) {
  // g * sigmoid(g) with v_rcp_f32 instead of the IEEE division sequence
  return g * __builtin_amdgcn_rcpf(1.f + __expf(-g));
}

// ---------------------------------------------------------------------------
__global__ void moe_init_kernel(int* counts) {
  if (threadIdx.x < E_NUM) counts[threadIdx.x] = 0;
}

// One wave per token: 8 logits, top-2, softmax over the 2 selected.
__global__ void moe_router_kernel(const float* __restrict__ x,
                                  const float* __restrict__ gw,
                                  int* __restrict__ topi, float* __restrict__ topp,
                                  int* __restrict__ counts) {
  __shared__ float red[8][32][E_NUM];
  const int wid = threadIdx.x >> 5, lane = threadIdx.x & 31;
  const int t = blockIdx.x * 8 + wid;
  float acc[E_NUM];
  #pragma unroll
  for (int e = 0; e < E_NUM; ++e) acc[e] = 0.f;
  for (int h = lane; h < H_DIM; h += 32) {
    float xv = x[t * H_DIM + h];
    const float* g = gw + h * E_NUM;
    #pragma unroll
    for (int e = 0; e < E_NUM; ++e) acc[e] += xv * g[e];
  }
  #pragma unroll
  for (int e = 0; e < E_NUM; ++e) red[wid][lane][e] = acc[e];
  __syncthreads();
  if (lane == 0) {
    float lg[E_NUM];
    for (int e = 0; e < E_NUM; ++e) {
      float s = 0.f;
      for (int l = 0; l < 32; ++l) s += red[wid][l][e];
      lg[e] = s;
    }
    int i0 = 0;
    for (int e = 1; e < E_NUM; ++e) if (lg[e] > lg[i0]) i0 = e;   // lowest index on tie
    int i1 = (i0 == 0) ? 1 : 0;
    for (int e = 0; e < E_NUM; ++e) if (e != i0 && lg[e] > lg[i1]) i1 = e;
    float p0 = 1.f / (1.f + __expf(lg[i1] - lg[i0]));
    topi[2 * t] = i0; topi[2 * t + 1] = i1;
    topp[2 * t] = p0; topp[2 * t + 1] = 1.f - p0;
    atomicAdd(&counts[i0], 1);
    atomicAdd(&counts[i1], 1);
  }
}

__global__ void moe_scan_kernel(const int* counts, int* offsets) {
  if (threadIdx.x == 0) {
    int s = 0;
    for (int e = 0; e < E_NUM; ++e) { offsets[e] = s; s += counts[e]; }
    offsets[E_NUM] = s;   // == 2*T_TOK always
  }
}

// One wave per expert: deterministic ordered slot lists via LDS prefix scan.
__global__ void moe_perm_kernel(const int* __restrict__ topi,
                                const float* __restrict__ topp,
                                const int* __restrict__ offsets,
                                int* __restrict__ perm, float* __restrict__ slotw,
                                int* __restrict__ tok2slot) {
  const int e = blockIdx.x, lane = threadIdx.x;
  __shared__ int flag[32];
  int base = offsets[e];
  for (int c = 0; c < T_TOK / 32; ++c) {
    int t = c * 32 + lane;
    int i0 = topi[2 * t], i1 = topi[2 * t + 1];
    int m = 0, j = 0; float w = 0.f;
    if (i0 == e)      { m = 1; j = 0; w = topp[2 * t]; }
    else if (i1 == e) { m = 1; j = 1; w = topp[2 * t + 1]; }
    flag[lane] = m;
    __syncthreads();
    int pre = 0, tot = 0;
    for (int l = 0; l < 32; ++l) { int f = flag[l]; tot += f; if (l < lane) pre += f; }
    if (m) {
      int s = base + pre;
      perm[s] = t; slotw[s] = w;
      tok2slot[2 * t + j] = s;
    }
    base += tot;
    __syncthreads();
  }
}

__global__ void moe_cvt_kernel(const float* __restrict__ x, bf16_t* __restrict__ xb, int n) {
  for (int i = blockIdx.x * blockDim.x + threadIdx.x; i < n; i += gridDim.x * blockDim.x)
    xb[i] = (bf16_t)x[i];
}

// ---------------------------------------------------------------------------
// Fused gate+up grouped GEMM + SiLU*up (+ routing weight folded into act).
// Block: 128 rows x 64 cols, 8 waves = 4x2, each wave 2x2 WMMA 16x16 tiles.
// K-step 32. Double-buffered LDS, single barrier per step.
// perm/offsets==nullptr -> identity rows (shared expert).
__global__ __launch_bounds__(256)
void moe_gateup_kernel(const bf16_t* __restrict__ X,
                       const float* __restrict__ Wg, const float* __restrict__ Wu,
                       long wstride,
                       const int* __restrict__ perm, const float* __restrict__ slotw,
                       const int* __restrict__ offsets, int Mfixed, int N,
                       bf16_t* __restrict__ act) {
  const int e = blockIdx.z;
  int base = 0, cnt = Mfixed;
  if (offsets) { base = offsets[e]; cnt = offsets[e + 1] - base; }
  const int m0 = blockIdx.y * 128;
  if (m0 >= cnt) return;
  const int n0 = blockIdx.x * 64;
  const float* wg = Wg + (long)e * wstride;
  const float* wu = Wu + (long)e * wstride;

  __shared__ alignas(16) bf16_t As[2][128 * 32];
  __shared__ alignas(16) bf16_t Bg[2][64 * 32];   // stored [n][k] (N-major)
  __shared__ alignas(16) bf16_t Bu[2][64 * 32];

  const int tid = threadIdx.x, wid = tid >> 5, lane = tid & 31;
  const int wm = wid >> 1, wn = wid & 1;

  v8f ag[2][2], au[2][2];
  #pragma unroll
  for (int i = 0; i < 2; ++i)
    #pragma unroll
    for (int j = 0; j < 2; ++j) { ag[i][j] = v8f_zero(); au[i][j] = v8f_zero(); }

  // A-tile loader: 2 threads per row, 16 bf16 (32B) each.
  const int arow = tid >> 1, ahalf = (tid & 1) * 16;
  int lrA = m0 + arow; if (lrA > cnt - 1) lrA = cnt - 1;   // clamp (dup rows never stored)
  const int tokA = perm ? perm[base + lrA] : (base + lrA);
  const bf16_t* aP = X + (long)tokA * H_DIM + ahalf;

  // B-tile loader: one (k-pair, n4) item per thread; packed b32 stores.
  const int k2 = tid >> 4;            // 0..15  -> k rows 2*k2, 2*k2+1
  const int ng = (tid & 15) * 4;      // 0..60
  const float* gP = wg + (long)(2 * k2) * N + n0 + ng;
  const float* uP = wu + (long)(2 * k2) * N + n0 + ng;
  const long bstep = (long)32 * N;

  const int r15 = lane & 15, kg = lane >> 4, ktop = kg * 16;

  uint4 sa0, sa1; float4 sg0, sg1, su0, su1;
  // prologue: stage tile 0
  sa0 = *reinterpret_cast<const uint4*>(aP);
  sa1 = *reinterpret_cast<const uint4*>(aP + 8);
  aP += 32;
  sg0 = *reinterpret_cast<const float4*>(gP);
  sg1 = *reinterpret_cast<const float4*>(gP + N);
  gP += bstep;
  su0 = *reinterpret_cast<const float4*>(uP);
  su1 = *reinterpret_cast<const float4*>(uP + N);
  uP += bstep;
  // commit tile 0 into buffer 0
  *reinterpret_cast<uint4*>(&As[0][arow * 32 + ahalf])     = sa0;
  *reinterpret_cast<uint4*>(&As[0][arow * 32 + ahalf + 8]) = sa1;
  #pragma unroll
  for (int j = 0; j < 4; ++j) {
    *reinterpret_cast<unsigned*>(&Bg[0][(ng + j) * 32 + 2 * k2]) =
        pack_bf16(reinterpret_cast<const float*>(&sg0)[j],
                  reinterpret_cast<const float*>(&sg1)[j]);
    *reinterpret_cast<unsigned*>(&Bu[0][(ng + j) * 32 + 2 * k2]) =
        pack_bf16(reinterpret_cast<const float*>(&su0)[j],
                  reinterpret_cast<const float*>(&su1)[j]);
  }

  const int niter = H_DIM / 32;
  int p = 0;
  for (int it = 0; it < niter; ++it) {
    __syncthreads();                       // buffer p ready; prior reads of p^1 done
    const bool more = (it + 1 < niter);
    if (more) {                            // issue next tile's global loads early
      sa0 = *reinterpret_cast<const uint4*>(aP);
      sa1 = *reinterpret_cast<const uint4*>(aP + 8);
      aP += 32;
      sg0 = *reinterpret_cast<const float4*>(gP);
      sg1 = *reinterpret_cast<const float4*>(gP + N);
      gP += bstep;
      su0 = *reinterpret_cast<const float4*>(uP);
      su1 = *reinterpret_cast<const float4*>(uP + N);
      uP += bstep;
    }

    // compute from buffer p (overlaps with the in-flight global loads)
    Frag16 fa[2], fg[2], fu[2];
    #pragma unroll
    for (int mi = 0; mi < 2; ++mi) {       // A 16x32 bf16: lane half h holds k = 16h..16h+15
      int row = wm * 32 + mi * 16 + r15;
      fa[mi].q[0] = *reinterpret_cast<const uint4*>(&As[p][row * 32 + ktop]);
      fa[mi].q[1] = *reinterpret_cast<const uint4*>(&As[p][row * 32 + ktop + 8]);
    }
    #pragma unroll
    for (int ni = 0; ni < 2; ++ni) {       // B 32x16: lane=col, K contiguous
      int col = wn * 32 + ni * 16 + r15;
      fg[ni].q[0] = *reinterpret_cast<const uint4*>(&Bg[p][col * 32 + ktop]);
      fg[ni].q[1] = *reinterpret_cast<const uint4*>(&Bg[p][col * 32 + ktop + 8]);
      fu[ni].q[0] = *reinterpret_cast<const uint4*>(&Bu[p][col * 32 + ktop]);
      fu[ni].q[1] = *reinterpret_cast<const uint4*>(&Bu[p][col * 32 + ktop + 8]);
    }
    #pragma unroll
    for (int mi = 0; mi < 2; ++mi)
      #pragma unroll
      for (int ni = 0; ni < 2; ++ni) {
        ag[mi][ni] = wmma_bf16(fa[mi], fg[ni], ag[mi][ni]);
        au[mi][ni] = wmma_bf16(fa[mi], fu[ni], au[mi][ni]);
      }

    if (more) {                            // commit staged tile into the other buffer
      int q = p ^ 1;
      *reinterpret_cast<uint4*>(&As[q][arow * 32 + ahalf])     = sa0;
      *reinterpret_cast<uint4*>(&As[q][arow * 32 + ahalf + 8]) = sa1;
      #pragma unroll
      for (int j = 0; j < 4; ++j) {
        *reinterpret_cast<unsigned*>(&Bg[q][(ng + j) * 32 + 2 * k2]) =
            pack_bf16(reinterpret_cast<const float*>(&sg0)[j],
                      reinterpret_cast<const float*>(&sg1)[j]);
        *reinterpret_cast<unsigned*>(&Bu[q][(ng + j) * 32 + 2 * k2]) =
            pack_bf16(reinterpret_cast<const float*>(&su0)[j],
                      reinterpret_cast<const float*>(&su1)[j]);
      }
    }
    p ^= 1;
  }

  // Epilogue: silu(g)*u * routing_weight -> bf16 act (slot-major).
  const int n15 = lane & 15, mb = (lane >> 4) * 8;
  #pragma unroll
  for (int mi = 0; mi < 2; ++mi)
    #pragma unroll
    for (int ni = 0; ni < 2; ++ni) {
      int col = n0 + wn * 32 + ni * 16 + n15;
      #pragma unroll
      for (int r = 0; r < 8; ++r) {
        int lrow = m0 + wm * 32 + mi * 16 + mb + r;
        if (lrow < cnt) {
          float sv = fast_silu(ag[mi][ni][r]) * au[mi][ni][r];
          float w = slotw ? slotw[base + lrow] : 1.f;
          act[(long)(base + lrow) * N + col] = (bf16_t)(sv * w);
        }
      }
    }
}

// Down-projection grouped GEMM. Rows are slot-contiguous (no gather).
// Plain f32 stores: shared path -> d_out (token rows), experts -> dbuf (slot rows).
__global__ __launch_bounds__(256)
void moe_down_kernel(const bf16_t* __restrict__ A, const float* __restrict__ Wd,
                     long wstride, const int* __restrict__ offsets, int Mfixed,
                     int KD, float* __restrict__ outbuf) {
  const int e = blockIdx.z;
  int base = 0, cnt = Mfixed;
  if (offsets) { base = offsets[e]; cnt = offsets[e + 1] - base; }
  const int m0 = blockIdx.y * 128;
  if (m0 >= cnt) return;
  const int n0 = blockIdx.x * 64;
  const float* wd = Wd + (long)e * wstride;

  __shared__ alignas(16) bf16_t As[2][128 * 32];
  __shared__ alignas(16) bf16_t Bs[2][64 * 32];

  const int tid = threadIdx.x, wid = tid >> 5, lane = tid & 31;
  const int wm = wid >> 1, wn = wid & 1;

  v8f acc[2][2];
  #pragma unroll
  for (int i = 0; i < 2; ++i)
    #pragma unroll
    for (int j = 0; j < 2; ++j) acc[i][j] = v8f_zero();

  const int arow = tid >> 1, ahalf = (tid & 1) * 16;
  int lrA = m0 + arow; if (lrA > cnt - 1) lrA = cnt - 1;
  const bf16_t* aP = A + (long)(base + lrA) * KD + ahalf;

  const int k2 = tid >> 4;
  const int ng = (tid & 15) * 4;
  const float* dP = wd + (long)(2 * k2) * H_DIM + n0 + ng;
  const long bstep = (long)32 * H_DIM;

  const int r15 = lane & 15, kg = lane >> 4, ktop = kg * 16;

  uint4 sa0, sa1; float4 sd0, sd1;
  sa0 = *reinterpret_cast<const uint4*>(aP);
  sa1 = *reinterpret_cast<const uint4*>(aP + 8);
  aP += 32;
  sd0 = *reinterpret_cast<const float4*>(dP);
  sd1 = *reinterpret_cast<const float4*>(dP + H_DIM);
  dP += bstep;
  *reinterpret_cast<uint4*>(&As[0][arow * 32 + ahalf])     = sa0;
  *reinterpret_cast<uint4*>(&As[0][arow * 32 + ahalf + 8]) = sa1;
  #pragma unroll
  for (int j = 0; j < 4; ++j)
    *reinterpret_cast<unsigned*>(&Bs[0][(ng + j) * 32 + 2 * k2]) =
        pack_bf16(reinterpret_cast<const float*>(&sd0)[j],
                  reinterpret_cast<const float*>(&sd1)[j]);

  const int niter = KD / 32;
  int p = 0;
  for (int it = 0; it < niter; ++it) {
    __syncthreads();
    const bool more = (it + 1 < niter);
    if (more) {
      sa0 = *reinterpret_cast<const uint4*>(aP);
      sa1 = *reinterpret_cast<const uint4*>(aP + 8);
      aP += 32;
      sd0 = *reinterpret_cast<const float4*>(dP);
      sd1 = *reinterpret_cast<const float4*>(dP + H_DIM);
      dP += bstep;
    }

    Frag16 fa[2], fb[2];
    #pragma unroll
    for (int mi = 0; mi < 2; ++mi) {
      int row = wm * 32 + mi * 16 + r15;
      fa[mi].q[0] = *reinterpret_cast<const uint4*>(&As[p][row * 32 + ktop]);
      fa[mi].q[1] = *reinterpret_cast<const uint4*>(&As[p][row * 32 + ktop + 8]);
    }
    #pragma unroll
    for (int ni = 0; ni < 2; ++ni) {
      int col = wn * 32 + ni * 16 + r15;
      fb[ni].q[0] = *reinterpret_cast<const uint4*>(&Bs[p][col * 32 + ktop]);
      fb[ni].q[1] = *reinterpret_cast<const uint4*>(&Bs[p][col * 32 + ktop + 8]);
    }
    #pragma unroll
    for (int mi = 0; mi < 2; ++mi)
      #pragma unroll
      for (int ni = 0; ni < 2; ++ni)
        acc[mi][ni] = wmma_bf16(fa[mi], fb[ni], acc[mi][ni]);

    if (more) {
      int q = p ^ 1;
      *reinterpret_cast<uint4*>(&As[q][arow * 32 + ahalf])     = sa0;
      *reinterpret_cast<uint4*>(&As[q][arow * 32 + ahalf + 8]) = sa1;
      #pragma unroll
      for (int j = 0; j < 4; ++j)
        *reinterpret_cast<unsigned*>(&Bs[q][(ng + j) * 32 + 2 * k2]) =
            pack_bf16(reinterpret_cast<const float*>(&sd0)[j],
                      reinterpret_cast<const float*>(&sd1)[j]);
    }
    p ^= 1;
  }

  const int n15 = lane & 15, mb = (lane >> 4) * 8;
  #pragma unroll
  for (int mi = 0; mi < 2; ++mi)
    #pragma unroll
    for (int ni = 0; ni < 2; ++ni) {
      int col = n0 + wn * 32 + ni * 16 + n15;
      #pragma unroll
      for (int r = 0; r < 8; ++r) {
        int lrow = m0 + wm * 32 + mi * 16 + mb + r;
        if (lrow < cnt)
          outbuf[(long)(base + lrow) * H_DIM + col] = acc[mi][ni][r];
      }
    }
}

// out[t] (holds shared-expert result) += dbuf[slot0] + dbuf[slot1]
__global__ void moe_combine_kernel(const int* __restrict__ tok2slot,
                                   const float* __restrict__ dbuf,
                                   float* __restrict__ out) {
  int t = blockIdx.x;
  int h = blockIdx.y * 256 + threadIdx.x;
  int s0 = tok2slot[2 * t], s1 = tok2slot[2 * t + 1];
  long o = (long)t * H_DIM + h;
  out[o] = out[o] + dbuf[(long)s0 * H_DIM + h] + dbuf[(long)s1 * H_DIM + h];
}

// ---------------------------------------------------------------------------
extern "C" void kernel_launch(void* const* d_in, const int* in_sizes, int n_in,
                              void* d_out, int out_size, void* d_ws, size_t ws_size,
                              hipStream_t stream) {
  const float* x  = (const float*)d_in[0];   // [2048,1024]
  const float* gw = (const float*)d_in[1];   // [1024,8]
  const float* Wg = (const float*)d_in[2];   // [8,1024,2816]
  const float* Wu = (const float*)d_in[3];
  const float* Wd = (const float*)d_in[4];   // [8,2816,1024]
  const float* Sg = (const float*)d_in[5];   // [1024,1408]
  const float* Su = (const float*)d_in[6];
  const float* Sd = (const float*)d_in[7];   // [1408,1024]
  float* out = (float*)d_out;                // [2048,1024]
  (void)in_sizes; (void)n_in; (void)out_size; (void)ws_size;

  char* ws = (char*)d_ws;
  size_t off = 0;
  auto take = [&](size_t bytes) -> char* {
    char* p = ws + off;
    off = (off + bytes + 255) & ~(size_t)255;
    return p;
  };
  int*    counts  = (int*)   take(E_NUM * sizeof(int));
  int*    offsets = (int*)   take((E_NUM + 1) * sizeof(int));
  int*    topi    = (int*)   take(2 * T_TOK * sizeof(int));
  float*  topp    = (float*) take(2 * T_TOK * sizeof(float));
  int*    perm    = (int*)   take(2 * T_TOK * sizeof(int));
  float*  slotw   = (float*) take(2 * T_TOK * sizeof(float));
  int*    t2s     = (int*)   take(2 * T_TOK * sizeof(int));
  bf16_t* xb      = (bf16_t*)take((size_t)T_TOK * H_DIM * sizeof(bf16_t));      // 4 MB
  bf16_t* act     = (bf16_t*)take((size_t)2 * T_TOK * I_DIM * sizeof(bf16_t));  // 23 MB
  bf16_t* sact    = (bf16_t*)take((size_t)T_TOK * IS_DIM * sizeof(bf16_t));     // 5.8 MB
  float*  dbuf    = (float*) take((size_t)2 * T_TOK * H_DIM * sizeof(float));   // 16.8 MB

  moe_init_kernel<<<1, 32, 0, stream>>>(counts);
  moe_router_kernel<<<T_TOK / 8, 256, 0, stream>>>(x, gw, topi, topp, counts);
  moe_scan_kernel<<<1, 1, 0, stream>>>(counts, offsets);
  moe_perm_kernel<<<E_NUM, 32, 0, stream>>>(topi, topp, offsets, perm, slotw, t2s);
  moe_cvt_kernel<<<2048, 256, 0, stream>>>(x, xb, T_TOK * H_DIM);

  // Shared expert (identity rows): writes d_out fully (plain stores).
  moe_gateup_kernel<<<dim3(IS_DIM / 64, T_TOK / 128, 1), 256, 0, stream>>>(
      xb, Sg, Su, 0L, nullptr, nullptr, nullptr, T_TOK, IS_DIM, sact);
  moe_down_kernel<<<dim3(H_DIM / 64, T_TOK / 128, 1), 256, 0, stream>>>(
      sact, Sd, 0L, nullptr, T_TOK, IS_DIM, out);

  // Routed experts: grouped GEMMs over slot lists (blocks past count early-exit).
  moe_gateup_kernel<<<dim3(I_DIM / 64, T_TOK / 128, E_NUM), 256, 0, stream>>>(
      xb, Wg, Wu, (long)H_DIM * I_DIM, perm, slotw, offsets, 0, I_DIM, act);
  moe_down_kernel<<<dim3(H_DIM / 64, T_TOK / 128, E_NUM), 256, 0, stream>>>(
      act, Wd, (long)I_DIM * H_DIM, offsets, 0, I_DIM, dbuf);

  moe_combine_kernel<<<dim3(T_TOK, H_DIM / 256), 256, 0, stream>>>(t2s, dbuf, out);
}
